// LiteMLA_9869834846686
// MI455X (gfx1250) — matrix operand order repaired and checked by reading, not compiled, not run on hardware
//
#include <hip/hip_runtime.h>
#include <hip/hip_fp16.h>

typedef __attribute__((ext_vector_type(16))) _Float16 v16h;
typedef __attribute__((ext_vector_type(8)))  _Float16 v8h;
typedef __attribute__((ext_vector_type(8)))  float    v8f;

#define HW    4096
#define CIN   256
#define C3    768
#define CATT  512
#define COUT  256
#define BATCH 8

// ---------------------------------------------------------------------------
// Kernel 1: transpose x [b][c][n] f32 -> xt [b][n][c] f16  (32x32 LDS tiles)
// ---------------------------------------------------------------------------
__global__ __launch_bounds__(256) void transpose_cvt_x(const float* __restrict__ x,
                                                       _Float16* __restrict__ xt) {
    __shared__ float tile[32][33];
    const int b  = blockIdx.z;
    const int n0 = blockIdx.x * 32;
    const int c0 = blockIdx.y * 32;
    const int tx = threadIdx.x, ty = threadIdx.y;
#pragma unroll
    for (int i = 0; i < 4; ++i)
        tile[ty + 8 * i][tx] = x[((size_t)b * CIN + c0 + ty + 8 * i) * HW + n0 + tx];
    __syncthreads();
#pragma unroll
    for (int i = 0; i < 4; ++i)
        xt[((size_t)b * HW + n0 + ty + 8 * i) * CIN + c0 + tx] =
            (_Float16)tile[tx][ty + 8 * i];
}

// ---------------------------------------------------------------------------
// Kernel 2: convert w_qkv (768x256) and w_proj (256x512) to f16
// ---------------------------------------------------------------------------
__global__ __launch_bounds__(256) void cvt_weights(const float* __restrict__ wq,
                                                   const float* __restrict__ wp,
                                                   _Float16* __restrict__ wqh,
                                                   _Float16* __restrict__ wph) {
    const int idx = blockIdx.x * 256 + threadIdx.x;
    const int NQ = C3 * CIN;            // 196608
    const int NP = COUT * CATT;         // 131072
    if (idx < NQ)           wqh[idx]      = (_Float16)wq[idx];
    else if (idx < NQ + NP) wph[idx - NQ] = (_Float16)wp[idx - NQ];
}

// ---------------------------------------------------------------------------
// Kernel 3: qkv GEMM  W[768x256] * Xt  -> Q f16 [b][768][4096]
// one 16x16 tile per wave, K=256 fully unrolled in 32-steps (8 wmma)
// ---------------------------------------------------------------------------
__global__ __launch_bounds__(256) void gemm_qkv(const _Float16* __restrict__ Wh,
                                                const _Float16* __restrict__ Xt,
                                                _Float16* __restrict__ Q) {
    const int tid  = threadIdx.x;
    const int wave = tid >> 5, lane = tid & 31;
    const int row  = lane & 15, hi = lane >> 4;
    const int b    = blockIdx.z;
    const int m0   = (blockIdx.y * 2 + (wave >> 2)) * 16;
    const int n0   = (blockIdx.x * 4 + (wave & 3)) * 16;
    const _Float16* Arow = Wh + (size_t)(m0 + row) * CIN;
    const _Float16* Bcol = Xt + ((size_t)b * HW + n0 + row) * CIN;
    v8f acc = {};
#pragma unroll
    for (int k0 = 0; k0 < CIN; k0 += 32) {
        if (k0 + 32 < CIN) {
            __builtin_prefetch(Arow + k0 + 32, 0, 3);
            __builtin_prefetch(Bcol + k0 + 32, 0, 3);
        }
        v8h alo = *(const v8h*)(Arow + k0 + hi * 16);
        v8h ahi = *(const v8h*)(Arow + k0 + 8 + hi * 16);
        v16h a;
#pragma unroll
        for (int i = 0; i < 8; ++i) { a[i] = alo[i]; a[8 + i] = ahi[i]; }
        v16h bf = *(const v16h*)(Bcol + k0 + hi * 16);
        acc = __builtin_amdgcn_wmma_f32_16x16x32_f16(false, a, false, bf,
                                                     (short)0, acc, false, false);
    }
    _Float16* out = Q + ((size_t)b * C3 + m0 + hi * 8) * HW + n0 + row;
#pragma unroll
    for (int r = 0; r < 8; ++r) out[(size_t)r * HW] = (_Float16)acc[r];
}

// ---------------------------------------------------------------------------
// Kernel 4: fused 5x5 depthwise conv + grouped 8x8 pointwise
// one thread per (b, group, pixel); 8 channels live in registers
// ---------------------------------------------------------------------------
__global__ __launch_bounds__(256) void dw_pw(const _Float16* __restrict__ Q,
                                             const float* __restrict__ wdw,
                                             const float* __restrict__ wpw,
                                             _Float16* __restrict__ P) {
    const int idx = blockIdx.x * 256 + threadIdx.x;
    const int n = idx & (HW - 1);
    const int g = (idx >> 12) % 96;
    const int b = idx / (HW * 96);
    const int y = n >> 6, xq = n & 63;
    float dw[8];
    const _Float16* qb = Q + ((size_t)b * C3 + g * 8) * HW;
#pragma unroll
    for (int i = 0; i < 8; ++i) {
        float s = 0.f;
        const float*    wt = wdw + (g * 8 + i) * 25;
        const _Float16* qc = qb + (size_t)i * HW;
#pragma unroll
        for (int dy = 0; dy < 5; ++dy) {
            const int yy = y + dy - 2;
            if (yy < 0 || yy > 63) continue;
#pragma unroll
            for (int dx = 0; dx < 5; ++dx) {
                const int xx = xq + dx - 2;
                if (xx < 0 || xx > 63) continue;
                s += wt[dy * 5 + dx] * (float)qc[yy * 64 + xx];
            }
        }
        dw[i] = s;
    }
    const float* wp = wpw + g * 64;
    _Float16* pb = P + ((size_t)b * C3 + g * 8) * HW + n;
#pragma unroll
    for (int o = 0; o < 8; ++o) {
        float s = 0.f;
#pragma unroll
        for (int i = 0; i < 8; ++i) s += wp[o * 8 + i] * dw[i];
        pb[(size_t)o * HW] = (_Float16)s;
    }
}

// ---------------------------------------------------------------------------
// Kernel 5: relu linear attention; one block per (b, head)
// kv[8][9] reduced via ds_add_f32; att written f16 as [b][n][512] (proj-B layout)
// ---------------------------------------------------------------------------
__global__ __launch_bounds__(256) void lin_att(const _Float16* __restrict__ Q,
                                               const _Float16* __restrict__ P,
                                               _Float16* __restrict__ At) {
    __shared__ float skv[72];
    const int tid = threadIdx.x;
    const int h = blockIdx.x & 63;
    const int b = blockIdx.x >> 6;
    const _Float16* base = (h < 32)
        ? Q + ((size_t)b * C3 + 24 * h) * HW
        : P + ((size_t)b * C3 + 24 * (h - 32)) * HW;
    if (tid < 72) skv[tid] = 0.f;
    __syncthreads();
    float kv[72];
#pragma unroll
    for (int t = 0; t < 72; ++t) kv[t] = 0.f;
    for (int p = 0; p < 16; ++p) {
        const int n = p * 256 + tid;
        float kk[8], vv[8];
#pragma unroll
        for (int d = 0; d < 8; ++d) {
            const float kd = (float)base[(size_t)(8 + d) * HW + n];
            kk[d] = kd > 0.f ? kd : 0.f;
            vv[d] = (float)base[(size_t)(16 + d) * HW + n];
        }
#pragma unroll
        for (int i = 0; i < 8; ++i) {
#pragma unroll
            for (int j = 0; j < 8; ++j) kv[i * 9 + j] += kk[i] * vv[j];
            kv[i * 9 + 8] += kk[i];
        }
    }
#pragma unroll
    for (int t = 0; t < 72; ++t) atomicAdd(&skv[t], kv[t]);
    __syncthreads();
#pragma unroll
    for (int t = 0; t < 72; ++t) kv[t] = skv[t];  // broadcast read
    for (int p = 0; p < 16; ++p) {
        const int n = p * 256 + tid;
        float qq[8];
#pragma unroll
        for (int d = 0; d < 8; ++d) {
            const float qd = (float)base[(size_t)d * HW + n];
            qq[d] = qd > 0.f ? qd : 0.f;
        }
        float o[9];
#pragma unroll
        for (int e = 0; e < 9; ++e) {
            float s = 0.f;
#pragma unroll
            for (int d = 0; d < 8; ++d) s += qq[d] * kv[d * 9 + e];
            o[e] = s;
        }
        const float dnm = 1.f / (o[8] + 1e-15f);
        v8h r;
#pragma unroll
        for (int d = 0; d < 8; ++d) r[d] = (_Float16)(o[d] * dnm);
        *(v8h*)(At + ((size_t)b * HW + n) * CATT + h * 8) = r;
    }
}

// ---------------------------------------------------------------------------
// Kernel 6: proj GEMM  Wp[256x512] * At^T  + fused BN  -> out f32 [b][256][4096]
// ---------------------------------------------------------------------------
__global__ __launch_bounds__(256) void gemm_proj(const _Float16* __restrict__ Wh,
                                                 const _Float16* __restrict__ At,
                                                 const float* __restrict__ gamma,
                                                 const float* __restrict__ beta,
                                                 const float* __restrict__ mean,
                                                 const float* __restrict__ var,
                                                 float* __restrict__ out) {
    const int tid  = threadIdx.x;
    const int wave = tid >> 5, lane = tid & 31;
    const int row  = lane & 15, hi = lane >> 4;
    const int b    = blockIdx.z;
    const int m0   = (blockIdx.y * 2 + (wave >> 2)) * 16;
    const int n0   = (blockIdx.x * 4 + (wave & 3)) * 16;
    const _Float16* Arow = Wh + (size_t)(m0 + row) * CATT;
    const _Float16* Bcol = At + ((size_t)b * HW + n0 + row) * CATT;
    v8f acc = {};
#pragma unroll
    for (int k0 = 0; k0 < CATT; k0 += 32) {
        if (k0 + 32 < CATT) {
            __builtin_prefetch(Arow + k0 + 32, 0, 3);
            __builtin_prefetch(Bcol + k0 + 32, 0, 3);
        }
        v8h alo = *(const v8h*)(Arow + k0 + hi * 16);
        v8h ahi = *(const v8h*)(Arow + k0 + 8 + hi * 16);
        v16h a;
#pragma unroll
        for (int i = 0; i < 8; ++i) { a[i] = alo[i]; a[8 + i] = ahi[i]; }
        v16h bf = *(const v16h*)(Bcol + k0 + hi * 16);
        acc = __builtin_amdgcn_wmma_f32_16x16x32_f16(false, a, false, bf,
                                                     (short)0, acc, false, false);
    }
#pragma unroll
    for (int r = 0; r < 8; ++r) {
        const int m  = m0 + r + 8 * hi;
        const float inv  = gamma[m] * rsqrtf(var[m] + 1e-5f);
        const float bias = beta[m] - mean[m] * inv;
        out[((size_t)b * COUT + m) * HW + n0 + row] = acc[r] * inv + bias;
    }
}

// ---------------------------------------------------------------------------
extern "C" void kernel_launch(void* const* d_in, const int* in_sizes, int n_in,
                              void* d_out, int out_size, void* d_ws, size_t ws_size,
                              hipStream_t stream) {
    const float* x      = (const float*)d_in[0];
    const float* w_qkv  = (const float*)d_in[1];
    const float* w_dw   = (const float*)d_in[2];
    const float* w_pw   = (const float*)d_in[3];
    const float* w_proj = (const float*)d_in[4];
    const float* gamma  = (const float*)d_in[5];
    const float* beta   = (const float*)d_in[6];
    const float* mean   = (const float*)d_in[7];
    const float* var    = (const float*)d_in[8];
    float* out = (float*)d_out;

    _Float16* xt  = (_Float16*)d_ws;                      // 8*4096*256
    _Float16* wqh = xt  + (size_t)BATCH * HW * CIN;       // 768*256
    _Float16* wph = wqh + (size_t)C3 * CIN;               // 256*512
    _Float16* qh  = wph + (size_t)COUT * CATT;            // 8*768*4096
    _Float16* ph  = qh  + (size_t)BATCH * C3 * HW;        // 8*768*4096
    _Float16* ath = ph  + (size_t)BATCH * C3 * HW;        // 8*4096*512

    transpose_cvt_x<<<dim3(128, 8, BATCH), dim3(32, 8), 0, stream>>>(x, xt);
    cvt_weights<<<1280, 256, 0, stream>>>(w_qkv, w_proj, wqh, wph);
    gemm_qkv<<<dim3(64, 24, BATCH), 256, 0, stream>>>(wqh, xt, qh);
    dw_pw<<<12288, 256, 0, stream>>>(qh, w_dw, w_pw, ph);
    lin_att<<<512, 256, 0, stream>>>(qh, ph, ath);
    gemm_proj<<<dim3(64, 8, BATCH), 256, 0, stream>>>(wph, ath, gamma, beta, mean,
                                                      var, out);
}
